// LabelPropGAT_84928683311554
// MI455X (gfx1250) — matrix operand order repaired and checked by reading, not compiled, not run on hardware
//
#include <hip/hip_runtime.h>
#include <hip/hip_bf16.h>

// ---------------------------------------------------------------------------
// Types for CDNA5 WMMA (wave32)
// ---------------------------------------------------------------------------
typedef __attribute__((ext_vector_type(16))) __bf16 v16bf;
typedef __attribute__((ext_vector_type(8)))  __bf16 v8bf;
typedef __attribute__((ext_vector_type(8)))  float  v8f;

#define LPG_N   50000
#define LPG_IN  256
#define LPG_F   128   // H*HID
#define LPG_C   64
#define LPG_SLOPE 0.2f

// ---------------------------------------------------------------------------
// Utility kernels
// ---------------------------------------------------------------------------
__global__ void lpg_fill(float* __restrict__ p, float v, long n) {
    long i = (long)blockIdx.x * blockDim.x + threadIdx.x;
    if (i < n) p[i] = v;
}

__global__ void lpg_cvt_bf16(const float* __restrict__ x, __bf16* __restrict__ y, long n) {
    long i = (long)blockIdx.x * blockDim.x + threadIdx.x;
    if (i < n) y[i] = (__bf16)x[i];
}

__global__ void lpg_elu_cvt_bf16(const float* __restrict__ x, __bf16* __restrict__ y, long n) {
    long i = (long)blockIdx.x * blockDim.x + threadIdx.x;
    if (i < n) {
        float v = x[i];
        v = v > 0.0f ? v : (__expf(v) - 1.0f);
        y[i] = (__bf16)v;
    }
}

// W [K, Nout] row-major f32  ->  Wt [Nout, K] row-major bf16
__global__ void lpg_transpose_cvt(const float* __restrict__ W, __bf16* __restrict__ Wt,
                                  int K, int Nout) {
    int i = blockIdx.x * blockDim.x + threadIdx.x;
    if (i >= K * Nout) return;
    int k = i / Nout, n = i % Nout;
    Wt[(long)n * K + k] = (__bf16)W[i];
}

__global__ void lpg_log(const float* __restrict__ x, float* __restrict__ y, long n) {
    long i = (long)blockIdx.x * blockDim.x + threadIdx.x;
    if (i < n) y[i] = __logf(fmaxf(x[i], 1e-30f));
}

// ---------------------------------------------------------------------------
// WMMA GEMM:  C[M,Nout] = A[M,K](bf16) * Bt[Nout,K](bf16)^T  (+ bias)
// grid.x = M/16 (M tile), blockDim.x = 32*(Nout/16); wave w -> N tile w.
// K multiple of 32; M multiple of 16.
// A per-lane layout (ISA 16-bit A 16x32): row = lane&15, two 8-elt chunks at
//   K = 8*(lane>>4) and K = 8*(lane>>4)+16.
// B per-lane layout (ISA 16-bit B 32x16): col = lane&15, two 8-elt chunks at
//   K = 8*(lane>>4) and K = 8*(lane>>4)+16.
// ---------------------------------------------------------------------------
__global__ void lpg_gemm_wmma(const __bf16* __restrict__ A,
                              const __bf16* __restrict__ Bt,
                              const float*  __restrict__ bias,
                              float* __restrict__ C,
                              int M, int K, int Nout) {
    const int lane = threadIdx.x & 31;
    const int wave = threadIdx.x >> 5;
    const int tm = blockIdx.x;
    const int tn = wave;

    const int arow  = tm * 16 + (lane & 15);
    const int bcol  = tn * 16 + (lane & 15);
    const int kA    = (lane >> 4) * 8;
    const int kB    = (lane >> 4) * 8;

    const __bf16* Ap = A  + (long)arow * K + kA;
    const __bf16* Bp = Bt + (long)bcol * K + kB;

    v8f acc = {};
    for (int k0 = 0; k0 < K; k0 += 32) {
        __builtin_prefetch(Ap + k0 + 64, 0, 1);   // global_prefetch_b8
        __builtin_prefetch(Bp + k0 + 64, 0, 1);
        v16bf a, b;
        *(v8bf*)&a       = *(const v8bf*)(Ap + k0);        // K: kA..kA+7
        *((v8bf*)&a + 1) = *(const v8bf*)(Ap + k0 + 16);   // K: kA+16..kA+23
        *(v8bf*)&b       = *(const v8bf*)(Bp + k0);        // K: kB..kB+7
        *((v8bf*)&b + 1) = *(const v8bf*)(Bp + k0 + 16);   // K: kB+16..kB+23
        acc = __builtin_amdgcn_wmma_f32_16x16x32_bf16(
            false, a, false, b, (short)0, acc, false, false);
    }

    // D layout: VGPR j -> (M = j + 8*(lane>>4), N = lane&15)
    const int col   = tn * 16 + (lane & 15);
    const int rbase = tm * 16 + (lane >> 4) * 8;
    const float bv  = bias ? bias[col] : 0.0f;
#pragma unroll
    for (int j = 0; j < 8; ++j)
        C[(long)(rbase + j) * Nout + col] = acc[j] + bv;
}

// ---------------------------------------------------------------------------
// GAT attention kernels
// ---------------------------------------------------------------------------
__device__ inline void lpg_atomicMaxF(float* addr, float val) {
    int* a = (int*)addr;
    int old = *a;
    while (__int_as_float(old) < val) {
        int assumed = old;
        old = atomicCAS(a, assumed, __float_as_int(val));
        if (old == assumed) break;
    }
}

// el[n,h] = sum_d h[n,h,d]*al[h,d]; er likewise. One thread per (n,h).
__global__ void lpg_node_scores(const float* __restrict__ h,
                                const float* __restrict__ al,
                                const float* __restrict__ ar,
                                float* __restrict__ el, float* __restrict__ er,
                                int N, int H, int D) {
    int i = blockIdx.x * blockDim.x + threadIdx.x;
    if (i >= N * H) return;
    int hh = i % H;
    const float* hp = h + (long)i * D;
    const float* alp = al + hh * D;
    const float* arp = ar + hh * D;
    float sl = 0.0f, sr = 0.0f;
    for (int d = 0; d < D; ++d) { float v = hp[d]; sl += v * alp[d]; sr += v * arp[d]; }
    el[i] = sl; er[i] = sr;
}

// e = leaky(el[src]+er[dst]); segment max into m[dst]. One thread per (e,h).
__global__ void lpg_edge_scores(const float* __restrict__ el, const float* __restrict__ er,
                                const int* __restrict__ src, const int* __restrict__ dst,
                                float* __restrict__ eout, float* __restrict__ m,
                                int E, int H) {
    int i = blockIdx.x * blockDim.x + threadIdx.x;
    if (i >= E * H) return;
    int e = i / H, hh = i % H;
    float v = el[src[e] * H + hh] + er[dst[e] * H + hh];
    v = v > 0.0f ? v : LPG_SLOPE * v;
    eout[i] = v;
    lpg_atomicMaxF(&m[dst[e] * H + hh], v);
}

// p = exp(e - m[dst]); segment sum into s[dst]. In-place over e buffer.
__global__ void lpg_edge_expsum(float* __restrict__ ep, const float* __restrict__ m,
                                const int* __restrict__ dst, float* __restrict__ s,
                                int E, int H) {
    int i = blockIdx.x * blockDim.x + threadIdx.x;
    if (i >= E * H) return;
    int e = i / H, hh = i % H;
    float p = __expf(ep[i] - m[dst[e] * H + hh]);
    ep[i] = p;
    atomicAdd(&s[dst[e] * H + hh], p);
}

// alpha = p / (s[dst]+1e-9) in place; wsum[e] += mean_h(alpha)/3. Thread/edge.
__global__ void lpg_edge_alpha(float* pa, const float* __restrict__ s,
                               const int* __restrict__ dst, float* __restrict__ wsum,
                               int E, int H) {
    int e = blockIdx.x * blockDim.x + threadIdx.x;
    if (e >= E) return;
    int d = dst[e];
    float acc = 0.0f;
    for (int hh = 0; hh < H; ++hh) {
        float a = pa[(long)e * H + hh] / (s[d * H + hh] + 1e-9f);
        pa[(long)e * H + hh] = a;
        acc += a;
    }
    wsum[e] += acc / ((float)H * 3.0f);
}

// out[dst,f] += h[src,f] * alpha[e, f/D]. One wave per edge, lanes stride F.
__global__ void lpg_gat_aggregate(const float* __restrict__ h, const float* alpha,
                                  const int* __restrict__ src, const int* __restrict__ dst,
                                  float* out, int E, int H, int D, int F) {
    int e = blockIdx.x * (blockDim.x >> 5) + (threadIdx.x >> 5);
    if (e >= E) return;
    int lane = threadIdx.x & 31;
    int sn = src[e], dn = dst[e];
    for (int f = lane; f < F; f += 32) {
        float a = alpha[(long)e * H + f / D];
        atomicAdd(&out[(long)dn * F + f], h[(long)sn * F + f] * a);
    }
}

// ---------------------------------------------------------------------------
// Softmax / label propagation
// ---------------------------------------------------------------------------
__global__ void lpg_softmax64(const float* __restrict__ x, float* __restrict__ y, int N) {
    int n = blockIdx.x * (blockDim.x >> 5) + (threadIdx.x >> 5);
    if (n >= N) return;
    int lane = threadIdx.x & 31;
    float v0 = x[(long)n * LPG_C + lane];
    float v1 = x[(long)n * LPG_C + lane + 32];
    float mx = fmaxf(v0, v1);
#pragma unroll
    for (int o = 16; o; o >>= 1) mx = fmaxf(mx, __shfl_xor(mx, o, 32));
    float e0 = __expf(v0 - mx), e1 = __expf(v1 - mx);
    float s = e0 + e1;
#pragma unroll
    for (int o = 16; o; o >>= 1) s += __shfl_xor(s, o, 32);
    float inv = 1.0f / s;
    y[(long)n * LPG_C + lane]      = e0 * inv;
    y[(long)n * LPG_C + lane + 32] = e1 * inv;
}

// agg[dst,c] += w[e] * pseudo[src,c]. One wave per edge, 2 elems/lane (C=64).
__global__ void lpg_lp_aggregate(const float* __restrict__ pseudo, const float* __restrict__ w,
                                 const int* __restrict__ src, const int* __restrict__ dst,
                                 float* agg, int E) {
    int e = blockIdx.x * (blockDim.x >> 5) + (threadIdx.x >> 5);
    if (e >= E) return;
    int lane = threadIdx.x & 31;
    int sn = src[e], dn = dst[e];
    float we = w[e];
    atomicAdd(&agg[(long)dn * LPG_C + lane],      we * pseudo[(long)sn * LPG_C + lane]);
    atomicAdd(&agg[(long)dn * LPG_C + lane + 32], we * pseudo[(long)sn * LPG_C + lane + 32]);
}

// ---------------------------------------------------------------------------
// Host orchestration
// ---------------------------------------------------------------------------
extern "C" void kernel_launch(void* const* d_in, const int* in_sizes, int n_in,
                              void* d_out, int out_size, void* d_ws, size_t ws_size,
                              hipStream_t stream) {
    const float* feat = (const float*)d_in[0];
    const int*   src  = (const int*)d_in[1];
    const int*   dst  = (const int*)d_in[2];
    const float* W0   = (const float*)d_in[3];
    const float* al0  = (const float*)d_in[4];
    const float* ar0  = (const float*)d_in[5];
    const float* W1   = (const float*)d_in[6];
    const float* al1  = (const float*)d_in[7];
    const float* ar1  = (const float*)d_in[8];
    const float* W2   = (const float*)d_in[9];
    const float* al2  = (const float*)d_in[10];
    const float* ar2  = (const float*)d_in[11];
    const float* Wp   = (const float*)d_in[12];
    const float* bp   = (const float*)d_in[13];

    const int N = LPG_N, IN = LPG_IN, F = LPG_F, C = LPG_C;
    const int E = in_sizes[1];

    // --- workspace carving ---
    char* ws = (char*)d_ws; size_t off = 0;
    auto carve = [&](size_t bytes) -> void* {
        void* p = ws + off; off = (off + bytes + 255) & ~(size_t)255; return p;
    };
    __bf16* featbf = (__bf16*)carve((size_t)N * IN * 2);
    __bf16* Wt0    = (__bf16*)carve((size_t)IN * F * 2);
    __bf16* Wt1    = (__bf16*)carve((size_t)F * F * 2);
    __bf16* Wt2    = (__bf16*)carve((size_t)F * C * 2);
    __bf16* Wtp    = (__bf16*)carve((size_t)C * C * 2);
    float*  hbuf   = (float*)carve((size_t)N * F * 4);   // transform out / logits
    float*  outbuf = (float*)carve((size_t)N * F * 4);   // aggregation out / agg
    __bf16* hinbf  = (__bf16*)carve((size_t)N * F * 2);  // bf16 layer input / agg_bf
    float*  el     = (float*)carve((size_t)N * 4 * 4);
    float*  er     = (float*)carve((size_t)N * 4 * 4);
    float*  mbuf   = (float*)carve((size_t)N * 4 * 4);
    float*  sbuf   = (float*)carve((size_t)N * 4 * 4);
    float*  ep     = (float*)carve((size_t)E * 4 * 4);   // e -> p -> alpha in place
    float*  wsum   = (float*)carve((size_t)E * 4);
    float*  pseudo = (float*)carve((size_t)N * C * 4);

    auto cdiv = [](long a, long b) { return (int)((a + b - 1) / b); };
    auto fill = [&](float* p, float v, long n) {
        lpg_fill<<<cdiv(n, 256), 256, 0, stream>>>(p, v, n);
    };

    // --- precision prep: bf16 features + transposed bf16 weights ---
    lpg_cvt_bf16<<<cdiv((long)N * IN, 256), 256, 0, stream>>>(feat, featbf, (long)N * IN);
    lpg_transpose_cvt<<<cdiv((long)IN * F, 256), 256, 0, stream>>>(W0, Wt0, IN, F);
    lpg_transpose_cvt<<<cdiv((long)F * F, 256), 256, 0, stream>>>(W1, Wt1, F, F);
    lpg_transpose_cvt<<<cdiv((long)F * C, 256), 256, 0, stream>>>(W2, Wt2, F, C);
    lpg_transpose_cvt<<<cdiv((long)C * C, 256), 256, 0, stream>>>(Wp, Wtp, C, C);
    fill(wsum, 0.0f, E);

    // --- one GAT layer ---
    auto gat_layer = [&](const __bf16* xin, const __bf16* Wt,
                         const float* alv, const float* arv,
                         int K, int H, int D) {
        const int Fl = H * D;
        lpg_gemm_wmma<<<N / 16, 32 * (Fl / 16), 0, stream>>>(xin, Wt, nullptr, hbuf, N, K, Fl);
        lpg_node_scores<<<cdiv((long)N * H, 256), 256, 0, stream>>>(hbuf, alv, arv, el, er, N, H, D);
        fill(mbuf, -1e30f, (long)N * H);
        fill(sbuf, 0.0f, (long)N * H);
        lpg_edge_scores<<<cdiv((long)E * H, 256), 256, 0, stream>>>(el, er, src, dst, ep, mbuf, E, H);
        lpg_edge_expsum<<<cdiv((long)E * H, 256), 256, 0, stream>>>(ep, mbuf, dst, sbuf, E, H);
        lpg_edge_alpha<<<cdiv(E, 256), 256, 0, stream>>>(ep, sbuf, dst, wsum, E, H);
        fill(outbuf, 0.0f, (long)N * Fl);
        lpg_gat_aggregate<<<cdiv(E, 8), 256, 0, stream>>>(hbuf, ep, src, dst, outbuf, E, H, D, Fl);
    };

    // Layer 0: 256 -> 4x32, ELU
    gat_layer(featbf, Wt0, al0, ar0, IN, 4, 32);
    lpg_elu_cvt_bf16<<<cdiv((long)N * F, 256), 256, 0, stream>>>(outbuf, hinbf, (long)N * F);
    // Layer 1: 128 -> 4x32, ELU
    gat_layer(hinbf, Wt1, al1, ar1, F, 4, 32);
    lpg_elu_cvt_bf16<<<cdiv((long)N * F, 256), 256, 0, stream>>>(outbuf, hinbf, (long)N * F);
    // Layer 2 (output): 128 -> 1x64; logits = head mean = outbuf itself
    gat_layer(hinbf, Wt2, al2, ar2, F, 1, 64);

    // pseudo = softmax(logits)
    lpg_softmax64<<<cdiv(N, 8), 256, 0, stream>>>(outbuf, pseudo, N);

    // --- label propagation: 10 steps ---
    for (int step = 0; step < 10; ++step) {
        fill(outbuf, 0.0f, (long)N * C);                                  // agg
        lpg_lp_aggregate<<<cdiv(E, 8), 256, 0, stream>>>(pseudo, wsum, src, dst, outbuf, E);
        lpg_cvt_bf16<<<cdiv((long)N * C, 256), 256, 0, stream>>>(outbuf, hinbf, (long)N * C);
        lpg_gemm_wmma<<<N / 16, 32 * (C / 16), 0, stream>>>(hinbf, Wtp, bp, hbuf, N, C, C);
        lpg_softmax64<<<cdiv(N, 8), 256, 0, stream>>>(hbuf, pseudo, N);
    }

    // output = log(pseudo)
    lpg_log<<<cdiv((long)N * C, 256), 256, 0, stream>>>(pseudo, (float*)d_out, (long)N * C);

    (void)n_in; (void)out_size; (void)ws_size;
}
